// DQN_21655225107236
// MI455X (gfx1250) — compile-verified, matrix operand order checked by reading, not run on hardware
//
#include <hip/hip_runtime.h>
#include <hip/hip_bf16.h>

typedef __attribute__((ext_vector_type(16))) _Float16 v16h;
typedef __attribute__((ext_vector_type(8)))  float    v8f;

// ---- WMMA helpers (CDNA5 V_WMMA_F32_16X16X32_F16 layouts, ISA 7.12.2) ----

// K index held by half-slot i (0..15) for a given lane, per the 16-bit A-matrix
// 16x32 layout: lanes 0-15 hold K {0..7, 16..23}; lanes 16-31 hold K {8..15, 24..31}.
__device__ __forceinline__ int kmap(int i, int lane) {
    int ko = (lane & 16) ? 8 : 0;
    return (i < 8) ? (ko + i) : (16 + ko + (i - 8));
}

__device__ __forceinline__ v8f wmma16(v16h a, v16h b, v8f c) {
    return __builtin_amdgcn_wmma_f32_16x16x32_f16(
        /*neg_a=*/false, a, /*neg_b=*/false, b,
        /*c_mod=*/(short)0, c, /*reuse_a=*/false, /*reuse_b=*/false);
}

// Pack B tile from an LDS-resident transposed weight W^T [Nout][K] (f16).
// col = global output column for this lane within the 16-wide tile.
__device__ __forceinline__ v16h pack_b_lds(const _Float16* wT, int col, int ldk,
                                           int kb, int lane) {
    v16h b;
#pragma unroll
    for (int i = 0; i < 16; ++i) b[i] = wT[col * ldk + kb * 32 + kmap(i, lane)];
    return b;
}

// ---- trivial data-movement kernels ----

__global__ void k_copy(const float* __restrict__ s, float* __restrict__ d, long n) {
    long i = (long)blockIdx.x * blockDim.x + threadIdx.x;
    if (i < n) d[i] = s[i];
}

__global__ void k_copy_x_to_z(const float* __restrict__ x, float* __restrict__ z, long n) {
    long i = (long)blockIdx.x * blockDim.x + threadIdx.x;
    if (i < n) z[(i >> 7) * 160 + (i & 127)] = x[i];  // z[:, 0:128] = x
}

// Edge-parallel segment_sum: one (edge, channel) per lane, coalesced per warp.
__global__ void k_scatter(const float* __restrict__ h, const int* __restrict__ src,
                          const int* __restrict__ dst, float* __restrict__ agg,
                          long total, int cshift) {
    long i = (long)blockIdx.x * blockDim.x + threadIdx.x;
    if (i >= total) return;
    int  C = 1 << cshift;
    long e = i >> cshift;
    int  c = (int)(i & (C - 1));
    int  sn = src[e], dn = dst[e];
    unsafeAtomicAdd(&agg[(long)dn * C + c], h[(long)sn * C + c]);
}

// ---- GIN MLP: out = relu(relu(in @ wa + ba) @ wb + bb), Nout=32, Kin in {32,128} ----

__global__ __launch_bounds__(128) void k_mlp(const float* __restrict__ in, int Kin, int Nn,
                                             const float* __restrict__ wa, const float* __restrict__ ba,
                                             const float* __restrict__ wb, const float* __restrict__ bb,
                                             float* __restrict__ out) {
    __shared__ _Float16 waT[32 * 128];   // wa^T, ldk = Kin
    __shared__ _Float16 wbT[32 * 32];    // wb^T, ldk = 32
    __shared__ float    sba[32], sbb[32];
    __shared__ _Float16 mid[4][16 * 32]; // per-wave staging (C-layout -> A-layout)

    int tid = threadIdx.x;
    for (int idx = tid; idx < Kin * 32; idx += 128) {
        int k = idx >> 5, n = idx & 31;
        waT[n * Kin + k] = (_Float16)wa[idx];          // wa[k][n], idx = k*32+n
    }
    for (int idx = tid; idx < 32 * 32; idx += 128) {
        int k = idx >> 5, n = idx & 31;
        wbT[n * 32 + k] = (_Float16)wb[idx];
    }
    if (tid < 32) { sba[tid] = ba[tid]; sbb[tid] = bb[tid]; }
    __syncthreads();

    int wave = tid >> 5, lane = tid & 31;
    int rowBase = blockIdx.x * 64 + wave * 16;
    int myRow   = rowBase + (lane & 15);
    int rdRow   = (myRow < Nn) ? myRow : (Nn - 1);     // keep EXEC all-ones
    const float* rowp = in + (long)rdRow * Kin;
    int col = lane & 15;

    v8f acc0 = {}, acc1 = {};
    int nkb = Kin >> 5;
    for (int kb = 0; kb < nkb; ++kb) {
        v16h a;
#pragma unroll
        for (int i = 0; i < 16; ++i) a[i] = (_Float16)rowp[kb * 32 + kmap(i, lane)];
        acc0 = wmma16(a, pack_b_lds(waT, col,      Kin, kb, lane), acc0);
        acc1 = wmma16(a, pack_b_lds(waT, col + 16, Kin, kb, lane), acc1);
    }
#pragma unroll
    for (int r = 0; r < 8; ++r) {
        int m = (lane < 16) ? r : r + 8;
        float v0 = acc0[r] + sba[col];      v0 = v0 > 0.f ? v0 : 0.f;
        float v1 = acc1[r] + sba[col + 16]; v1 = v1 > 0.f ? v1 : 0.f;
        mid[wave][m * 32 + col]      = (_Float16)v0;
        mid[wave][m * 32 + col + 16] = (_Float16)v1;
    }
    __syncthreads();

    v16h a2;
#pragma unroll
    for (int i = 0; i < 16; ++i) a2[i] = mid[wave][(lane & 15) * 32 + kmap(i, lane)];
    v8f zero = {};
    v8f o0 = wmma16(a2, pack_b_lds(wbT, col,      32, 0, lane), zero);
    v8f o1 = wmma16(a2, pack_b_lds(wbT, col + 16, 32, 0, lane), zero);
#pragma unroll
    for (int r = 0; r < 8; ++r) {
        int m = (lane < 16) ? r : r + 8;
        int g = rowBase + m;
        if (g < Nn) {
            float v0 = fmaxf(o0[r] + sbb[col], 0.f);       // outer relu of GIN layer
            float v1 = fmaxf(o1[r] + sbb[col + 16], 0.f);
            out[(long)g * 32 + col]      = v0;
            out[(long)g * 32 + col + 16] = v1;
        }
    }
}

// ---- jk='cat' projection: z[:,128:160] = [h1|h2|h3] @ jk_w + jk_b (3 K-blocks) ----

__global__ __launch_bounds__(128) void k_jk(const float* __restrict__ h1,
                                            const float* __restrict__ h2,
                                            const float* __restrict__ h3,
                                            const float* __restrict__ jkw,
                                            const float* __restrict__ jkb,
                                            float* __restrict__ z, int Nn) {
    __shared__ _Float16 wT[32 * 96];     // jk_w^T, ldk = 96
    __shared__ float    sb[32];
    int tid = threadIdx.x;
    for (int idx = tid; idx < 96 * 32; idx += 128) {
        int k = idx >> 5, n = idx & 31;
        wT[n * 96 + k] = (_Float16)jkw[idx];
    }
    if (tid < 32) sb[tid] = jkb[tid];
    __syncthreads();

    int wave = tid >> 5, lane = tid & 31;
    int rowBase = blockIdx.x * 64 + wave * 16;
    int myRow   = rowBase + (lane & 15);
    int rdRow   = (myRow < Nn) ? myRow : (Nn - 1);
    int col = lane & 15;

    v8f acc0 = {}, acc1 = {};
#pragma unroll
    for (int kb = 0; kb < 3; ++kb) {
        const float* hp = (kb == 0 ? h1 : (kb == 1 ? h2 : h3)) + (long)rdRow * 32;
        v16h a;
#pragma unroll
        for (int i = 0; i < 16; ++i) a[i] = (_Float16)hp[kmap(i, lane)];
        acc0 = wmma16(a, pack_b_lds(wT, col,      96, kb, lane), acc0);
        acc1 = wmma16(a, pack_b_lds(wT, col + 16, 96, kb, lane), acc1);
    }
#pragma unroll
    for (int r = 0; r < 8; ++r) {
        int m = (lane < 16) ? r : r + 8;
        int g = rowBase + m;
        if (g < Nn) {
            z[(long)g * 160 + 128 + col]      = acc0[r] + sb[col];
            z[(long)g * 160 + 128 + col + 16] = acc1[r] + sb[col + 16];
        }
    }
}

// ---- row L2-normalize z [N,160]: one wave per node ----

__global__ __launch_bounds__(32) void k_normalize(float* __restrict__ z, int Nn) {
    int n = blockIdx.x;
    if (n >= Nn) return;
    int lane = threadIdx.x;
    float s = 0.f;
    for (int j = lane; j < 160; j += 32) { float v = z[(long)n * 160 + j]; s += v * v; }
#pragma unroll
    for (int off = 16; off; off >>= 1) s += __shfl_xor(s, off, 32);
    float inv = rsqrtf(s);
    for (int j = lane; j < 160; j += 32) z[(long)n * 160 + j] *= inv;
}

// ---- pair MLP: relu(comb@o0+b)@... ; comb built on the fly from gathered z rows ----
// Tile = 16 pairs sharing one src row, 16 consecutive dst rows. 256 tiles/graph.

__global__ __launch_bounds__(128) void k_pairs(const float* __restrict__ z,
                                               const int* __restrict__ src_idx,
                                               const int* __restrict__ dst_idx,
                                               const unsigned char* __restrict__ s_mask,
                                               const unsigned char* __restrict__ d_mask,
                                               const float* __restrict__ o0w, const float* __restrict__ o0b,
                                               const float* __restrict__ o1w, const float* __restrict__ o1b,
                                               const float* __restrict__ o2w, const float* __restrict__ o2b,
                                               float* __restrict__ out) {
    __shared__ _Float16 w0T[64 * 160];   // o0_w^T [64][160]
    __shared__ _Float16 w1T[32 * 64];    // o1_w^T [32][64]
    __shared__ float    sb0[64], sb1[32], sw2[32];
    __shared__ _Float16 mid1[4][16 * 64];
    __shared__ float    mid2[4][16 * 32];

    int tid = threadIdx.x;
    for (int idx = tid; idx < 160 * 64; idx += 128) {
        int k = idx >> 6, n = idx & 63;
        w0T[n * 160 + k] = (_Float16)o0w[idx];
    }
    for (int idx = tid; idx < 64 * 32; idx += 128) {
        int k = idx >> 5, n = idx & 31;
        w1T[n * 64 + k] = (_Float16)o1w[idx];
    }
    if (tid < 64) sb0[tid] = o0b[tid];
    if (tid < 32) { sb1[tid] = o1b[tid]; sw2[tid] = o2w[tid]; }
    __syncthreads();

    int wave = tid >> 5, lane = tid & 31;
    int tg = blockIdx.x * 4 + wave;      // global tile id
    int b  = tg >> 8;                    // 256 tiles per graph
    int t  = tg & 255;
    int s  = t >> 2;                     // src slot 0..63
    int d0 = (t & 3) << 4;               // dst base 0,16,32,48
    int col = lane & 15;

    int srow = src_idx[b * 64 + s];
    int drow = dst_idx[b * 64 + d0 + col];   // row m of the tile uses dst d0+m
    const float* sz = z + (long)srow * 160;
    const float* dz = z + (long)drow * 160;

    // GEMM1: [16 x 160] x [160 x 64]
    v8f acc[4] = {};
    for (int kb = 0; kb < 5; ++kb) {
        v16h a;
#pragma unroll
        for (int i = 0; i < 16; ++i) {
            int k = kb * 32 + kmap(i, lane);
            a[i] = (_Float16)(sz[k] * dz[k]);
        }
#pragma unroll
        for (int nt = 0; nt < 4; ++nt)
            acc[nt] = wmma16(a, pack_b_lds(w0T, nt * 16 + col, 160, kb, lane), acc[nt]);
    }
#pragma unroll
    for (int nt = 0; nt < 4; ++nt) {
        int gc = nt * 16 + col;
#pragma unroll
        for (int r = 0; r < 8; ++r) {
            int m = (lane < 16) ? r : r + 8;
            float v = acc[nt][r] + sb0[gc];
            mid1[wave][m * 64 + gc] = (_Float16)(v > 0.f ? v : 0.f);
        }
    }
    __syncthreads();

    // GEMM2: [16 x 64] x [64 x 32]
    v8f acc2[2] = {};
    for (int kb = 0; kb < 2; ++kb) {
        v16h a;
#pragma unroll
        for (int i = 0; i < 16; ++i)
            a[i] = mid1[wave][(lane & 15) * 64 + kb * 32 + kmap(i, lane)];
#pragma unroll
        for (int nt = 0; nt < 2; ++nt)
            acc2[nt] = wmma16(a, pack_b_lds(w1T, nt * 16 + col, 64, kb, lane), acc2[nt]);
    }
#pragma unroll
    for (int nt = 0; nt < 2; ++nt) {
        int gc = nt * 16 + col;
#pragma unroll
        for (int r = 0; r < 8; ++r) {
            int m = (lane < 16) ? r : r + 8;
            float v = acc2[nt][r] + sb1[gc];
            mid2[wave][m * 32 + gc] = (v > 0.f ? v : 0.f);
        }
    }
    __syncthreads();

    // GEMM3: dot with o2_w [32] -> scalar per pair; lanes split K in halves.
    int row  = lane & 15;
    int half = lane >> 4;
    float sum = 0.f;
#pragma unroll
    for (int k = 0; k < 16; ++k) {
        int kk = half * 16 + k;
        sum += mid2[wave][row * 32 + kk] * sw2[kk];
    }
    sum += __shfl_xor(sum, 16, 32);
    if (half == 0) {
        int dd = d0 + row;
        float v = sum + o2b[0];
        bool masked = (s_mask[b * 64 + s] != 0) || (d_mask[b * 64 + dd] != 0);
        out[(long)b * 4096 + s * 64 + dd] = masked ? -__builtin_inff() : v;
    }
}

// ---- host orchestration ----

extern "C" void kernel_launch(void* const* d_in, const int* in_sizes, int n_in,
                              void* d_out, int out_size, void* d_ws, size_t ws_size,
                              hipStream_t stream) {
    const float* x        = (const float*)d_in[0];
    const int*   eidx     = (const int*)d_in[1];
    const int*   src_idx  = (const int*)d_in[2];
    const int*   dst_idx  = (const int*)d_in[3];
    const unsigned char* s_mask = (const unsigned char*)d_in[4];
    const unsigned char* d_mask = (const unsigned char*)d_in[5];
    const float* w0a = (const float*)d_in[6],  *b0a = (const float*)d_in[7];
    const float* w0b = (const float*)d_in[8],  *b0b = (const float*)d_in[9];
    const float* w1a = (const float*)d_in[10], *b1a = (const float*)d_in[11];
    const float* w1b = (const float*)d_in[12], *b1b = (const float*)d_in[13];
    const float* w2a = (const float*)d_in[14], *b2a = (const float*)d_in[15];
    const float* w2b = (const float*)d_in[16], *b2b = (const float*)d_in[17];
    const float* jkw = (const float*)d_in[18], *jkb = (const float*)d_in[19];
    const float* o0w = (const float*)d_in[20], *o0b = (const float*)d_in[21];
    const float* o1w = (const float*)d_in[22], *o1b = (const float*)d_in[23];
    const float* o2w = (const float*)d_in[24], *o2b = (const float*)d_in[25];

    const int  N = in_sizes[0] / 128;
    const long E = in_sizes[1] / 2;
    const int  B = in_sizes[2] / 64;
    const int* esrc = eidx;
    const int* edst = eidx + E;

    float* agg0 = (float*)d_ws;                        // [N,128]
    float* h1   = agg0 + (size_t)N * 128;              // [N,32]
    float* h2   = h1   + (size_t)N * 32;               // [N,32]
    float* h3   = h2   + (size_t)N * 32;               // [N,32]
    float* aggS = h3   + (size_t)N * 32;               // [N,32] (reused)
    float* zbuf = aggS + (size_t)N * 32;               // [N,160]

    const int THR = 256;
    long n128 = (long)N * 128, n32 = (long)N * 32;
    int nodeBlocks = (N + 63) / 64;

    // Layer 0: agg = x + segment_sum(x[src], dst); h1 = GIN MLP
    k_copy<<<(int)((n128 + THR - 1) / THR), THR, 0, stream>>>(x, agg0, n128);
    k_scatter<<<(int)((E * 128 + THR - 1) / THR), THR, 0, stream>>>(x, esrc, edst, agg0, E * 128, 7);
    k_mlp<<<nodeBlocks, 128, 0, stream>>>(agg0, 128, N, w0a, b0a, w0b, b0b, h1);

    // Layer 1
    k_copy<<<(int)((n32 + THR - 1) / THR), THR, 0, stream>>>(h1, aggS, n32);
    k_scatter<<<(int)((E * 32 + THR - 1) / THR), THR, 0, stream>>>(h1, esrc, edst, aggS, E * 32, 5);
    k_mlp<<<nodeBlocks, 128, 0, stream>>>(aggS, 32, N, w1a, b1a, w1b, b1b, h2);

    // Layer 2
    k_copy<<<(int)((n32 + THR - 1) / THR), THR, 0, stream>>>(h2, aggS, n32);
    k_scatter<<<(int)((E * 32 + THR - 1) / THR), THR, 0, stream>>>(h2, esrc, edst, aggS, E * 32, 5);
    k_mlp<<<nodeBlocks, 128, 0, stream>>>(aggS, 32, N, w2a, b2a, w2b, b2b, h3);

    // z = normalize([x | jk([h1,h2,h3])])
    k_copy_x_to_z<<<(int)((n128 + THR - 1) / THR), THR, 0, stream>>>(x, zbuf, n128);
    k_jk<<<nodeBlocks, 128, 0, stream>>>(h1, h2, h3, jkw, jkb, zbuf, N);
    k_normalize<<<N, 32, 0, stream>>>(zbuf, N);

    // all-pairs scoring MLP
    k_pairs<<<B * 64, 128, 0, stream>>>(zbuf, src_idx, dst_idx, s_mask, d_mask,
                                        o0w, o0b, o1w, o1b, o2w, o2b, (float*)d_out);
}
